// DeepEMD_Metric_23656679866495
// MI455X (gfx1250) — matrix operand order repaired and checked by reading, not compiled, not run on hardware
//
#include <hip/hip_runtime.h>

// ---------------------------------------------------------------------------
// DeepEMD metric for MI455X (gfx1250, wave32, WMMA).
// Pipeline:
//   P1 proto_prep   : x2 -> proto (shot mean), pooled_p, pn bf16 [way][c][208]
//   P2 query_prep   : x1 -> pooled_q, qn bf16 [m][208][c] (LDS-tiled transpose)
//   P3 weights_prep : loga/logb marginals [m][way][196]
//   MAIN emd_main   : per (m,way) workgroup: WMMA bf16 GEMM -> sim in LDS,
//                     50 log-Sinkhorn iterations in LDS, logit reduction.
// Round-2 change: software-pipelined A/B fragment loads in the WMMA loop so
// global loads overlap WMMA issue instead of load->wait(0)->wmma serialization.
// ---------------------------------------------------------------------------

typedef __bf16 bf16;
typedef __bf16 v16bf __attribute__((ext_vector_type(16)));
typedef __bf16 v8bf  __attribute__((ext_vector_type(8)));
typedef float  v8f   __attribute__((ext_vector_type(8)));

#define M_Q   200
#define WAY   5
#define C_DIM 640
#define N_SP  196      // 14*14
#define N_PAD 208      // 13*16
#define NT    13       // 16x16 tiles per dim
#define SIM_STRIDE 197 // conflict-free LDS stride (5*row mod 64 banking)

// ---------------- block-wide sum reduction (wave32 shfl + LDS) -------------
__device__ inline float block_reduce_sum(float v, float* red) {
  #pragma unroll
  for (int off = 16; off > 0; off >>= 1) v += __shfl_xor(v, off, 32);
  const int wave = threadIdx.x >> 5;
  const int lane = threadIdx.x & 31;
  if (lane == 0) red[wave] = v;
  __syncthreads();
  const int nw = blockDim.x >> 5;
  float s = (threadIdx.x < nw) ? red[threadIdx.x] : 0.f;
  #pragma unroll
  for (int off = 16; off > 0; off >>= 1) s += __shfl_xor(s, off, 32);
  if (threadIdx.x == 0) red[0] = s;
  __syncthreads();
  float out = red[0];
  __syncthreads();                 // safe for immediate reuse of red[]
  return out;
}

// ---------------- P1: proto (shot mean) -> pooled_p, pn bf16 ---------------
__global__ void proto_prep(const float* __restrict__ x2,
                           float* __restrict__ proto,
                           float* __restrict__ pooled_p,
                           bf16* __restrict__ pn) {
  const int way = blockIdx.x;
  const int a   = threadIdx.x;          // 256 threads, active a < 196 (/208)
  __shared__ float red[16];
  __shared__ float s_mean[N_SP], s_inv[N_SP];

  float csum = 0.f, cssq = 0.f;
  for (int c = 0; c < C_DIM; ++c) {
    float v = 0.f;
    if (a < N_SP) {
      #pragma unroll
      for (int s = 0; s < 5; ++s)
        v += x2[(((way * 5 + s) * C_DIM) + c) * N_SP + a];
      v *= 0.2f;
      proto[(way * C_DIM + c) * N_SP + a] = v;
    }
    csum += v; cssq += v * v;
    float tot = block_reduce_sum(v, red);
    if (threadIdx.x == 0) pooled_p[way * C_DIM + c] = tot * (1.f / (float)N_SP);
  }
  if (a < N_SP) {
    float mean = csum * (1.f / (float)C_DIM);
    float ssq  = fmaxf(cssq - (float)C_DIM * mean * mean, 0.f);
    s_mean[a] = mean;
    s_inv[a]  = 1.f / fmaxf(sqrtf(ssq), 1e-8f);
  }
  __syncthreads();
  // pn layout: [way][c][a(pad 208)]  (K-major => contiguous B-fragment rows)
  for (int c = 0; c < C_DIM; ++c) {
    if (a < N_SP) {
      float v = proto[(way * C_DIM + c) * N_SP + a];
      pn[(way * C_DIM + c) * N_PAD + a] = (bf16)((v - s_mean[a]) * s_inv[a]);
    } else if (a < N_PAD) {
      pn[(way * C_DIM + c) * N_PAD + a] = (bf16)0.f;
    }
  }
}

// ---------------- P2: query -> pooled_q, qn bf16 [m][a(pad)][c] ------------
__global__ void query_prep(const float* __restrict__ x1,
                           float* __restrict__ pooled_q,
                           bf16* __restrict__ qn) {
  const int m = blockIdx.x;
  const int a = threadIdx.x;            // 256 threads
  __shared__ float red[16];
  __shared__ float s_mean[N_SP], s_inv[N_SP];
  __shared__ bf16  tile[32][200];       // 32 channels x 196(+pad) positions

  float csum = 0.f, cssq = 0.f;
  for (int c = 0; c < C_DIM; ++c) {
    float v = (a < N_SP) ? x1[(m * C_DIM + c) * N_SP + a] : 0.f;
    csum += v; cssq += v * v;
    float tot = block_reduce_sum(v, red);
    if (threadIdx.x == 0) pooled_q[m * C_DIM + c] = tot * (1.f / (float)N_SP);
  }
  if (a < N_SP) {
    float mean = csum * (1.f / (float)C_DIM);
    float ssq  = fmaxf(cssq - (float)C_DIM * mean * mean, 0.f);
    s_mean[a] = mean;
    s_inv[a]  = 1.f / fmaxf(sqrtf(ssq), 1e-8f);
  }
  __syncthreads();

  // Transpose [c][a] -> [a][c] through LDS so qn rows are contiguous in c.
  const int tr = threadIdx.x >> 5;      // 0..7
  const int tc = threadIdx.x & 31;      // 0..31
  for (int c0 = 0; c0 < C_DIM; c0 += 32) {
    for (int cc = 0; cc < 32; ++cc) {
      if (a < N_SP)
        tile[cc][a] = (bf16)((x1[(m * C_DIM + c0 + cc) * N_SP + a] - s_mean[a]) * s_inv[a]);
    }
    __syncthreads();
    for (int a0 = 0; a0 < N_PAD; a0 += 8) {
      int ar = a0 + tr;
      bf16 v = (ar < N_SP) ? tile[tc][ar] : (bf16)0.f;   // zero pad rows 196..207
      qn[((size_t)m * N_PAD + ar) * C_DIM + c0 + tc] = v;
    }
    __syncthreads();
  }
}

// ---------------- P3: Sinkhorn marginals loga/logb -------------------------
__global__ void weights_prep(const float* __restrict__ x1,
                             const float* __restrict__ proto,
                             const float* __restrict__ pooled_p,
                             const float* __restrict__ pooled_q,
                             float* __restrict__ loga,
                             float* __restrict__ logb) {
  const int m = blockIdx.x, way = blockIdx.y;
  __shared__ float red[16];
  __shared__ float sp[C_DIM], sq[C_DIM];
  for (int i = threadIdx.x; i < C_DIM; i += blockDim.x) {
    sp[i] = pooled_p[way * C_DIM + i];
    sq[i] = pooled_q[m * C_DIM + i];
  }
  __syncthreads();
  const int a = threadIdx.x;
  float acc1 = 0.f, acc2 = 0.f;
  if (a < N_SP) {
    for (int c = 0; c < C_DIM; ++c) {
      acc1 += x1[(m * C_DIM + c) * N_SP + a] * sp[c];
      acc2 += proto[(way * C_DIM + c) * N_SP + a] * sq[c];
    }
  }
  float w1 = (a < N_SP) ? fmaxf(acc1, 0.f) + 1e-3f : 0.f;
  float w2 = (a < N_SP) ? fmaxf(acc2, 0.f) + 1e-3f : 0.f;
  float s1 = block_reduce_sum(w1, red);
  float s2 = block_reduce_sum(w2, red);
  if (a < N_SP) {
    const int base = (m * WAY + way) * N_SP;
    loga[base + a] = logf(w1) - logf(s1);
    logb[base + a] = logf(w2) - logf(s2);   // b = w2^T normalized over spatial
  }
}

// ---------------- MAIN: WMMA sim -> LDS, Sinkhorn, logits ------------------
__global__ __launch_bounds__(416, 1)
void emd_main(const bf16* __restrict__ qn, const bf16* __restrict__ pn,
              const float* __restrict__ loga, const float* __restrict__ logb,
              float* __restrict__ out) {
  const int way = blockIdx.x;   // 0..4
  const int m   = blockIdx.y;   // 0..199

  extern __shared__ float smem[];
  float* s_sim  = smem;                          // 196 * 197
  float* s_loga = smem + N_SP * SIM_STRIDE;      // 196
  float* s_logb = s_loga + N_SP;                 // 196
  float* s_f    = s_logb + N_SP;                 // 196
  float* s_g    = s_f + N_SP;                    // 196
  float* s_red  = s_g + N_SP;                    // 16

  const int tid  = threadIdx.x;
  const int wave = tid >> 5;    // 0..12 -> output row tile
  const int lane = tid & 31;

  // preload marginals, init potentials
  {
    const int base = (m * WAY + way) * N_SP;
    for (int i = tid; i < N_SP; i += 416) {
      s_loga[i] = loga[base + i];
      s_logb[i] = logb[base + i];
      s_f[i] = 0.f;
      s_g[i] = 0.f;
    }
  }

  // ---- Phase 1: sim[196,196] = qn[m]^T(196x640) @ pn[way](640x196), bf16 WMMA
  v8f acc[NT];
  {
    const v8f zero = {0.f,0.f,0.f,0.f,0.f,0.f,0.f,0.f};
    #pragma unroll
    for (int nt = 0; nt < NT; ++nt) acc[nt] = zero;
  }
  const bf16* Abase = qn + (size_t)m * (N_PAD * C_DIM);
  const bf16* Bbase = pn + (size_t)way * (C_DIM * N_PAD);
  const int half = lane >> 4;        // A-fragment K-half select
  const int arow = lane & 15;        // A-fragment M row within tile

  // A fragment (16x32 bf16): lane<16 holds K {0..7,16..23}, lane>=16 {8..15,24..31}
  auto loadA = [&](int k0) -> v16bf {
    const bf16* pa = Abase + (wave * 16 + arow) * C_DIM + k0 + half * 8;
    v8bf lo = *(const v8bf*)(pa);
    v8bf hi = *(const v8bf*)(pa + 16);
    v16bf o;
    #pragma unroll
    for (int i = 0; i < 8; ++i) { o[i] = lo[i]; o[i + 8] = hi[i]; }
    return o;
  };
  // B fragment (32x16 bf16): lane holds column N=lane&15, K {0..7,16..23} (lane<16) / {8..15,24..31}
  auto loadB = [&](int k0, int nt) -> v16bf {
    const bf16* pb = Bbase + (size_t)(k0 + half * 8) * N_PAD + nt * 16 + arow;
    v16bf o;
    #pragma unroll
    for (int i = 0; i < 8; ++i) { o[i] = pb[i * N_PAD]; o[i + 8] = pb[(i + 16) * N_PAD]; }
    return o;
  };

  // Software pipeline: the load feeding WMMA step t+1 is issued before the
  // WMMA of step t, so global-load latency overlaps matrix issue.
  v16bf afrag = loadA(0);
  v16bf bfrag = loadB(0, 0);
  for (int k0 = 0; k0 < C_DIM; k0 += 32) {
    v16bf anext = afrag;
    #pragma unroll
    for (int nt = 0; nt < NT; ++nt) {
      v16bf bnext;
      if (nt < NT - 1) {
        bnext = loadB(k0, nt + 1);
      } else {
        const int kn = (k0 + 32 < C_DIM) ? (k0 + 32) : k0;  // clamped tail reload
        bnext = loadB(kn, 0);
        anext = loadA(kn);
      }
      acc[nt] = __builtin_amdgcn_wmma_f32_16x16x32_bf16(
          false, afrag, false, bfrag, (short)0, acc[nt], false, false);
      bfrag = bnext;
    }
    afrag = anext;
  }

  // store accumulators into LDS sim (C/D layout: vgpr v -> M=v+8*(lane>>4), N=lane&15)
  {
    const int lh = lane >> 4, ln = lane & 15;
    #pragma unroll
    for (int nt = 0; nt < NT; ++nt) {
      const int col = nt * 16 + ln;
      if (col < N_SP) {
        #pragma unroll
        for (int v = 0; v < 8; ++v) {
          const int row = wave * 16 + v + 8 * lh;
          if (row < N_SP) s_sim[row * SIM_STRIDE + col] = acc[nt][v];
        }
      }
    }
  }
  __syncthreads();

  // ---- Phase 2: log-domain Sinkhorn, eps=0.05, 50 iterations
  const float EPS = 0.05f, INV_EPS = 20.0f;
  const float L2E = 1.4426950408889634f, LN2 = 0.6931471805599453f;
  const float CEXP = INV_EPS * L2E;     // exp(x/eps) == exp2(x*CEXP)

  const int r   = tid >> 1;             // row/col handled by a thread pair
  const int h   = tid & 1;              // which half of the 196-long sweep
  const bool act = (r < N_SP);
  const int rr  = act ? r : 0;
  const int j0  = h * 98, j1 = j0 + 98;

  for (int it = 0; it < 50; ++it) {
    // f_i = eps*(loga_i - LSE_j((g_j - 1 + sim_ij)/eps)); max over t = g_j + sim_ij
    {
      const float* simrow = s_sim + rr * SIM_STRIDE;
      float mx = -1e30f;
      for (int j = j0; j < j1; ++j) mx = fmaxf(mx, s_g[j] + simrow[j]);
      float s = 0.f;
      for (int j = j0; j < j1; ++j) s += exp2f((s_g[j] + simrow[j] - mx) * CEXP);
      float mxo = __shfl_xor(mx, 1, 32);
      float so  = __shfl_xor(s, 1, 32);
      float Mv = fmaxf(mx, mxo);
      float Sv = s * exp2f((mx - Mv) * CEXP) + so * exp2f((mxo - Mv) * CEXP);
      float lse = (Mv - 1.0f) * INV_EPS + log2f(Sv) * LN2;
      if (act && h == 0) s_f[rr] = EPS * s_loga[rr] - EPS * lse;
    }
    __syncthreads();
    // g_j = eps*(logb_j - LSE_i((f_i - 1 + sim_ij)/eps))
    {
      float mx = -1e30f;
      for (int i = j0; i < j1; ++i) mx = fmaxf(mx, s_f[i] + s_sim[i * SIM_STRIDE + rr]);
      float s = 0.f;
      for (int i = j0; i < j1; ++i) s += exp2f((s_f[i] + s_sim[i * SIM_STRIDE + rr] - mx) * CEXP);
      float mxo = __shfl_xor(mx, 1, 32);
      float so  = __shfl_xor(s, 1, 32);
      float Mv = fmaxf(mx, mxo);
      float Sv = s * exp2f((mx - Mv) * CEXP) + so * exp2f((mxo - Mv) * CEXP);
      float lse = (Mv - 1.0f) * INV_EPS + log2f(Sv) * LN2;
      if (act && h == 0) s_g[rr] = EPS * s_logb[rr] - EPS * lse;
    }
    __syncthreads();
  }

  // ---- Phase 3: logit = T * sum_ij sim_ij * exp((f_i + g_j - 1 + sim_ij)/eps)
  float accv = 0.f;
  if (act) {
    const float* simrow = s_sim + rr * SIM_STRIDE;
    const float fr = s_f[rr];
    for (int j = j0; j < j1; ++j) {
      float sv = simrow[j];
      accv += sv * exp2f((fr + s_g[j] - 1.0f + sv) * CEXP);
    }
  }
  float tot = block_reduce_sum(accv, s_red);
  if (tid == 0) out[m * WAY + way] = tot * 12.5f;
}

// ---------------------------------------------------------------------------
extern "C" void kernel_launch(void* const* d_in, const int* in_sizes, int n_in,
                              void* d_out, int out_size, void* d_ws, size_t ws_size,
                              hipStream_t stream) {
  (void)in_sizes; (void)n_in; (void)out_size; (void)ws_size;
  const float* x1 = (const float*)d_in[0];   // [200,640,14,14]
  const float* x2 = (const float*)d_in[1];   // [25,640,14,14]
  float* out = (float*)d_out;                // [200,5]

  char* ws = (char*)d_ws;
  float* proto    = (float*)(ws);                 // 5*640*196*4      = 2,508,800
  float* pooled_p = (float*)(ws +  2508800);      // 5*640*4          =    12,800
  float* pooled_q = (float*)(ws +  2521600);      // 200*640*4        =   512,000
  bf16*  pn       = (bf16*) (ws +  3033600);      // 5*640*208*2      = 1,331,200
  bf16*  qn       = (bf16*) (ws +  4364800);      // 200*208*640*2    = 53,248,000
  float* loga     = (float*)(ws + 57612800);      // 200*5*196*4      =   784,000
  float* logb     = (float*)(ws + 58396800);      // 200*5*196*4      =   784,000

  proto_prep  <<<WAY, 256, 0, stream>>>(x2, proto, pooled_p, pn);
  query_prep  <<<M_Q, 256, 0, stream>>>(x1, pooled_q, qn);
  weights_prep<<<dim3(M_Q, WAY), 256, 0, stream>>>(x1, proto, pooled_p, pooled_q, loga, logb);

  const size_t smem_bytes = (size_t)(N_SP * SIM_STRIDE + 4 * N_SP + 16) * sizeof(float); // 157,648 B
  emd_main<<<dim3(WAY, M_Q), 416, smem_bytes, stream>>>(qn, pn, loga, logb, out);
}
